// TopKSparseAutoencoder_56255481643318
// MI455X (gfx1250) — matrix operand order repaired and checked by reading, not compiled, not run on hardware
//
#include <hip/hip_runtime.h>

// ---- CDNA5 WMMA types ------------------------------------------------------
typedef __attribute__((ext_vector_type(16))) __bf16 v16bf;
typedef __attribute__((ext_vector_type(2)))  __bf16 v2bf;
typedef __attribute__((ext_vector_type(8)))  float  v8f;

// ---- problem sizes ---------------------------------------------------------
#define T_TOKENS 4096          // B*N = 16*256
#define D_CONCAT 4096
#define D_HIDDEN 16384
#define D_INPUT  1024
#define KTOP     64

// ---- GEMM tiling -----------------------------------------------------------
#define BM 128
#define BN 128
#define BK 64                   // two 16x16x32 k-steps per stage

// fp32 pair -> packed bf16 (RNE). Prefer the single-instruction packed
// convert; fall back to rounding-add + one v_perm_b32 byte shuffle.
__device__ __forceinline__ unsigned pack_bf16(float a, float b) {
#if __has_builtin(__builtin_amdgcn_cvt_pk_bf16_f32)
    v2bf r = __builtin_amdgcn_cvt_pk_bf16_f32(a, b);   // lo=a, hi=b
    return __builtin_bit_cast(unsigned, r);
#else
    unsigned ua = __float_as_uint(a), ub = __float_as_uint(b);
    ua += 0x7FFFu + ((ua >> 16) & 1u);
    ub += 0x7FFFu + ((ub >> 16) & 1u);
    // bytes: dst = { ub[3], ub[2], ua[3], ua[2] }  (src1=ua -> bytes 0-3)
    return __builtin_amdgcn_perm(ub, ua, 0x07060302u);
#endif
}

union FragAB { v16bf v; uint4 q[2]; };

// C[M][NTOT] = op( A'[M][KTOT] * B[NTOT][KTOT]^T + bias[NTOT] )
//   A' = A - a_center (broadcast along K) when SUB_A_BIAS
//   op = relu when RELU;  DEC_OUT also scatters the 4 layer_rec views.
// Ping-pong LDS staging (1 barrier / 16 WMMAs) + register prefetch of the
// next K-tile: global-load latency is covered by the current tile's WMMAs,
// LDS-commit of stage k reads registers fetched one full stage earlier.
template<int KTOT, int NTOT, bool SUB_A_BIAS, bool RELU, bool DEC_OUT>
__global__ __launch_bounds__(256)
void sae_gemm(const float* __restrict__ A,
              const float* __restrict__ B,
              const float* __restrict__ a_center,
              const float* __restrict__ n_bias,
              float* __restrict__ C,
              float* __restrict__ Clayer)
{
    __shared__ __align__(16) unsigned lsA[2][BM * BK / 2];   // 2 x 16 KB
    __shared__ __align__(16) unsigned lsB[2][BN * BK / 2];   // 2 x 16 KB

    const int tid    = threadIdx.x;
    const int lane   = tid & 31;
    const int wid    = tid >> 5;       // 8 waves
    const int wave_m = wid & 3;        // 4 wave-rows * 32 rows
    const int wave_n = wid >> 2;       // 2 wave-cols * 64 cols
    const int bm = blockIdx.y * BM;
    const int bn = blockIdx.x * BN;

    const int r16 = lane & 15;
    const int hi  = lane >> 4;

    const v8f vzero = {0.f,0.f,0.f,0.f,0.f,0.f,0.f,0.f};
    v8f acc[2][4];
    #pragma unroll
    for (int i = 0; i < 2; ++i)
        #pragma unroll
        for (int j = 0; j < 4; ++j) acc[i][j] = vzero;

    // per-thread staging slots: 2048 float4 per tile / 256 threads = 8 each
    const int sr = tid >> 4;            // base row; rows advance by 16 per i
    const int sc = (tid & 15) << 2;     // col 0..60 step 4

    float4 pa[8], pb[8];
    // ---- prologue: prefetch K-tile 0 --------------------------------------
    #pragma unroll
    for (int i = 0; i < 8; ++i) {
        int r = sr + i * 16;
        float4 av = *(const float4*)(A + (size_t)(bm + r) * KTOT + sc);
        if (SUB_A_BIAS) {
            float4 s = *(const float4*)(a_center + sc);
            av.x -= s.x; av.y -= s.y; av.z -= s.z; av.w -= s.w;
        }
        pa[i] = av;
        pb[i] = *(const float4*)(B + (size_t)(bn + r) * KTOT + sc);
    }

    const int nIter = KTOT / BK;
    for (int it = 0; it < nIter; ++it) {
        unsigned* dA = lsA[it & 1];
        unsigned* dB = lsB[it & 1];

        // ---- commit prefetched tile to LDS (fp32 -> bf16) ----
        #pragma unroll
        for (int i = 0; i < 8; ++i) {
            int r = sr + i * 16;
            uint2 ua, ub;
            ua.x = pack_bf16(pa[i].x, pa[i].y);
            ua.y = pack_bf16(pa[i].z, pa[i].w);
            ub.x = pack_bf16(pb[i].x, pb[i].y);
            ub.y = pack_bf16(pb[i].z, pb[i].w);
            *(uint2*)&dA[r * 32 + (sc >> 1)] = ua;   // row stride 64 bf16
            *(uint2*)&dB[r * 32 + (sc >> 1)] = ub;
        }
        __syncthreads();   // single barrier per stage (ping-pong buffers)

        // ---- prefetch next K-tile; latency hidden behind the WMMAs below ----
        const int kn = (it + 1) * BK;
        if (kn < KTOT) {
            #pragma unroll
            for (int i = 0; i < 8; ++i) {
                int r = sr + i * 16;
                float4 av = *(const float4*)(A + (size_t)(bm + r) * KTOT + kn + sc);
                if (SUB_A_BIAS) {
                    float4 s = *(const float4*)(a_center + kn + sc);
                    av.x -= s.x; av.y -= s.y; av.z -= s.z; av.w -= s.w;
                }
                pa[i] = av;
                pb[i] = *(const float4*)(B + (size_t)(bn + r) * KTOT + kn + sc);
            }
        }

        const uint4* qA = (const uint4*)dA;   // uint4 == 16B == 8 bf16
        const uint4* qB = (const uint4*)dB;

        #pragma unroll
        for (int kk = 0; kk < 2; ++kk) {      // two 16x16x32 k-steps
            // A frag: lane L<16 -> row L, K{0..7,16..23}; L>=16 -> K{8..15,24..31}
            FragAB afr[2];
            #pragma unroll
            for (int mt = 0; mt < 2; ++mt) {
                int row = wave_m * 32 + mt * 16 + r16;
                afr[mt].q[0] = qA[row * 8 + kk * 4 + hi];
                afr[mt].q[1] = qA[row * 8 + kk * 4 + 2 + hi];
            }
            // B frag: lane L<16 -> col L, K{0..7,16..23}; L>=16 -> K{8..15,24..31}
            FragAB bfr[4];
            #pragma unroll
            for (int nt = 0; nt < 4; ++nt) {
                int col = wave_n * 64 + nt * 16 + r16;
                bfr[nt].q[0] = qB[col * 8 + kk * 4 + hi];
                bfr[nt].q[1] = qB[col * 8 + kk * 4 + 2 + hi];
            }
            #pragma unroll
            for (int mt = 0; mt < 2; ++mt)
                #pragma unroll
                for (int nt = 0; nt < 4; ++nt)
                    acc[mt][nt] = __builtin_amdgcn_wmma_f32_16x16x32_bf16(
                        false, afr[mt].v, false, bfr[nt].v,
                        (short)0, acc[mt][nt], false, false);
        }
    }

    // ---- epilogue: C/D layout VGPR r, lane L -> M = r + 8*(L>=16), N = L&15 ----
    #pragma unroll
    for (int mt = 0; mt < 2; ++mt) {
        #pragma unroll
        for (int nt = 0; nt < 4; ++nt) {
            int col    = bn + wave_n * 64 + nt * 16 + r16;
            float bias = n_bias[col];
            int row0   = bm + wave_m * 32 + mt * 16 + hi * 8;
            #pragma unroll
            for (int r = 0; r < 8; ++r) {
                float v = acc[mt][nt][r] + bias;
                if (RELU) v = fmaxf(v, 0.0f);
                int row = row0 + r;
                C[(size_t)row * NTOT + col] = v;
                if (DEC_OUT) {
                    int l = col >> 10;          // layer index
                    int d = col & 1023;         // within-layer feature
                    Clayer[(size_t)l * ((size_t)T_TOKENS * D_INPUT)
                           + (size_t)row * D_INPUT + d] = v;
                }
            }
        }
    }
}

// Exact top-64 per 16384-wide row via 4x8-bit radix select on float bits
// (valid: all values >= 0 after ReLU, so uint order == float order).
// Zeroes non-kept entries in place.
__global__ __launch_bounds__(256)
void topk64_kernel(float* __restrict__ hs)
{
    const int row = blockIdx.x;
    const int tid = threadIdx.x;
    float* rptr = hs + (size_t)row * D_HIDDEN;

    unsigned vals[D_HIDDEN / 256];      // 64 elems per thread
    #pragma unroll 8
    for (int j = 0; j < D_HIDDEN / 256; ++j)
        vals[j] = __float_as_uint(rptr[tid + j * 256]);

    __shared__ unsigned hist[256];
    __shared__ unsigned s_prefix, s_remaining, s_tie;
    if (tid == 0) { s_prefix = 0u; s_remaining = KTOP; s_tie = 0u; }

    for (int p = 0; p < 4; ++p) {
        const int shift = 24 - 8 * p;
        hist[tid] = 0u;
        __syncthreads();
        const unsigned pref = s_prefix;
        for (int j = 0; j < D_HIDDEN / 256; ++j) {
            unsigned u = vals[j];
            bool cand = (p == 0) || ((u >> (shift + 8)) == pref);
            if (cand) atomicAdd(&hist[(u >> shift) & 255u], 1u);
        }
        __syncthreads();
        if (tid == 0) {
            unsigned rem = s_remaining, c = 0; int sel = 0;
            for (int b = 255; b >= 0; --b) {
                unsigned h = hist[b];
                if (c + h >= rem) { sel = b; break; }
                c += h;
            }
            s_remaining = rem - c;                    // ties at threshold to keep
            s_prefix    = (pref << 8) | (unsigned)sel;
        }
        __syncthreads();
    }

    const unsigned thr       = s_prefix;
    const unsigned keep_ties = s_remaining;
    for (int j = 0; j < D_HIDDEN / 256; ++j) {
        unsigned u = vals[j];
        bool keep;
        if (u > thr)       keep = true;
        else if (u == thr) keep = (atomicAdd(&s_tie, 1u) < keep_ties);
        else               keep = false;
        if (!keep) rptr[tid + j * 256] = 0.0f;
    }
}

extern "C" void kernel_launch(void* const* d_in, const int* in_sizes, int n_in,
                              void* d_out, int out_size, void* d_ws, size_t ws_size,
                              hipStream_t stream)
{
    (void)in_sizes; (void)n_in; (void)out_size; (void)d_ws; (void)ws_size;

    const float* z     = (const float*)d_in[0];
    const float* W_enc = (const float*)d_in[1];
    const float* b_enc = (const float*)d_in[2];
    const float* W_dec = (const float*)d_in[3];
    const float* b_dec = (const float*)d_in[4];

    float* out    = (float*)d_out;
    float* z_rec  = out;                                        // (4096, 4096)
    float* hs     = out + (size_t)T_TOKENS * D_CONCAT;          // (4096, 16384)
    float* layers = hs  + (size_t)T_TOKENS * D_HIDDEN;          // 4 x (4096, 1024)

    // 1) h = relu((z - b_dec) @ W_enc^T + b_enc), dense into h_sparse region
    sae_gemm<D_CONCAT, D_HIDDEN, true, true, false>
        <<<dim3(D_HIDDEN / BN, T_TOKENS / BM), 256, 0, stream>>>(
            z, W_enc, b_dec, b_enc, hs, nullptr);

    // 2) in-place top-64 sparsification per token row
    topk64_kernel<<<T_TOKENS, 256, 0, stream>>>(hs);

    // 3) z_rec = h_sparse @ W_dec^T + b_dec, fused layer_rec scatter
    sae_gemm<D_HIDDEN, D_CONCAT, false, false, true>
        <<<dim3(D_CONCAT / BN, T_TOKENS / BM), 256, 0, stream>>>(
            hs, W_dec, nullptr, b_dec, z_rec, layers);
}
